// RelativeBiasMultiHeadAttention_3599182594648
// MI455X (gfx1250) — matrix operand order repaired and checked by reading, not compiled, not run on hardware
//
#include <hip/hip_runtime.h>

// ---------------------------------------------------------------------------
// RelativeBiasMultiHeadAttention for MI455X (gfx1250), bf16 WMMA path.
// B=2, S=2048, D=1024, H=16, DK=64, MAX_SEQ=2048.
// Requires ws_size >= 64 MB (67,108,864 bytes).
// ---------------------------------------------------------------------------

typedef __bf16 bf16;
typedef __attribute__((ext_vector_type(16))) __bf16 v16bf;
typedef __attribute__((ext_vector_type(8)))  __bf16 v8bf;
typedef __attribute__((ext_vector_type(4)))  __bf16 v4bf;
typedef __attribute__((ext_vector_type(8)))  float  v8f;
typedef __attribute__((ext_vector_type(4)))  float  v4f;

constexpr int Bx      = 2;
constexpr int Sx      = 2048;
constexpr int Dx      = 1024;
constexpr int Hx      = 16;
constexpr int DKx     = 64;
constexpr int MAXSEQ  = 2048;

// ---------------------------------------------------------------------------
// CDNA5 async global->LDS copies (GLOBAL_LOAD_ASYNC_TO_LDS_B128, ASYNCcnt).
// Probe round 2 confirmed the builtin exists with param 1 being
// 'int __attribute__((vector_size(16))) addrspace(1)*' (global src first).
// Guarded so the file still compiles if a toolchain lacks the builtins.
// ---------------------------------------------------------------------------
#if __has_builtin(__builtin_amdgcn_global_load_async_to_lds_b128) && \
    __has_builtin(__builtin_amdgcn_s_wait_asynccnt)
#define HAVE_ASYNC_LDS 1
#else
#define HAVE_ASYNC_LDS 0
#endif

#if HAVE_ASYNC_LDS
typedef int v4i_vs __attribute__((vector_size(16)));
typedef __attribute__((address_space(1))) v4i_vs gas_v4i;   // global int4
typedef __attribute__((address_space(3))) v4i_vs las_v4i;   // LDS int4
__device__ inline void cp_async_b128(bf16* lds_dst, const bf16* g_src) {
  __builtin_amdgcn_global_load_async_to_lds_b128(
      (gas_v4i*)g_src, (las_v4i*)lds_dst, /*offset=*/0, /*cpol=*/0);
}
__device__ inline void cp_async_wait0() {
  __builtin_amdgcn_s_wait_asynccnt(0);
}
#endif

// ---------------------------------------------------------------------------
// WMMA fragment helpers (layouts per CDNA5 ISA §7.12.2, wave32)
// ---------------------------------------------------------------------------

// A-matrix 16x32 bf16 from row-major tile (ld in elements).
// lanes 0-15: M=lane,   elems 0-7 -> K=0..7,   elems 8-15 -> K=16..23
// lanes16-31: M=lane-16,elems 0-7 -> K=8..15,  elems 8-15 -> K=24..31
__device__ inline v16bf load_frag_A(const bf16* __restrict__ tile, int ld) {
  const int lane = threadIdx.x & 31;
  const int half = lane >> 4;
  const int m    = lane & 15;
  const bf16* row = tile + (size_t)m * ld + 8 * half;
  v8bf c0 = *(const v8bf*)(row);        // K = 8h .. 8h+7
  v8bf c1 = *(const v8bf*)(row + 16);   // K = 8h+16 .. 8h+23
  v16bf a;
#pragma unroll
  for (int i = 0; i < 8; ++i) { a[i] = c0[i]; a[8 + i] = c1[i]; }
  return a;
}

// B-matrix 32x16 bf16 where the source is row-major [N][K] (B[k][n]=src[n][k]).
// lanes 0-15: N=lane, K=0..7,16..23 ; lanes16-31: N=lane-16, K=8..15,24..31.
__device__ inline v16bf load_frag_B_nk(const bf16* __restrict__ tile, int ld) {
  const int lane = threadIdx.x & 31;
  const int half = lane >> 4;
  const int n    = lane & 15;
  const bf16* row = tile + (size_t)n * ld + 8 * half;
  v8bf c0 = *(const v8bf*)(row);
  v8bf c1 = *(const v8bf*)(row + 16);
  v16bf b;
#pragma unroll
  for (int i = 0; i < 8; ++i) { b[i] = c0[i]; b[8 + i] = c1[i]; }
  return b;
}

__device__ inline v8f wmma_bf16(v16bf a, v16bf b, v8f c) {
  return __builtin_amdgcn_wmma_f32_16x16x32_bf16(
      /*neg_a=*/false, a, /*neg_b=*/false, b,
      /*c_mod=*/(short)0, c, /*reuse_a=*/false, /*reuse_b=*/false);
}

// ---------------------------------------------------------------------------
// f32 -> bf16 conversion (grid-stride)
// ---------------------------------------------------------------------------
__global__ void f32_to_bf16_kernel(const float* __restrict__ src,
                                   bf16* __restrict__ dst, size_t n) {
  size_t i = (size_t)blockIdx.x * blockDim.x + threadIdx.x;
  size_t stride = (size_t)gridDim.x * blockDim.x;
  for (; i < n; i += stride) dst[i] = (bf16)src[i];
}

// ---------------------------------------------------------------------------
// Generic bf16 GEMM:  C[m,n] = sum_k A[m,k] * W[n,k]   (both row-major, ld=K)
// mode 0: out f32 row-major [M,N], plus bias[n] if bias != nullptr
// mode 1: out bf16 head-split   [B,H,S,DK]   (m=(b,s), n=(h,dk))
// mode 2: out bf16 head-split-T [B,H,DK,S]
// Block: 256 threads (8 waves), tile BM=64 x BN=128, BK=64; wave tile 32x32.
// Double-buffered LDS pipeline: async copies if available, else register
// prefetch of the next K-tile overlapped with the WMMAs of the current one.
// ---------------------------------------------------------------------------
#define GEMM_BM 64
#define GEMM_BN 128
#define GEMM_BK 64

__global__ __launch_bounds__(256)
void gemm_bf16_nt(const bf16* __restrict__ A, const bf16* __restrict__ W,
                  const float* __restrict__ bias, float* __restrict__ outF,
                  bf16* __restrict__ outB, int M, int N, int K, int mode) {
  __shared__ alignas(16) bf16 ldsA[2][GEMM_BM * GEMM_BK];   // 2 x 8 KB
  __shared__ alignas(16) bf16 ldsB[2][GEMM_BN * GEMM_BK];   // 2 x 16 KB, [n][k]

  const int tid  = threadIdx.x;
  const int wave = tid >> 5;
  const int wm   = wave & 1;          // 2 waves over M (32 rows each)
  const int wn   = wave >> 1;         // 4 waves over N (32 cols each)
  const int m0   = blockIdx.y * GEMM_BM;
  const int n0   = blockIdx.x * GEMM_BN;

  v8f acc[2][2] = {};

  auto compute = [&](int buf) {
#pragma unroll
    for (int ks = 0; ks < GEMM_BK; ks += 32) {
      v16bf af[2], bfr[2];
#pragma unroll
      for (int i = 0; i < 2; ++i)
        af[i] = load_frag_A(&ldsA[buf][(wm * 32 + i * 16) * GEMM_BK + ks],
                            GEMM_BK);
#pragma unroll
      for (int j = 0; j < 2; ++j)
        bfr[j] = load_frag_B_nk(&ldsB[buf][(wn * 32 + j * 16) * GEMM_BK + ks],
                                GEMM_BK);
#pragma unroll
      for (int i = 0; i < 2; ++i)
#pragma unroll
        for (int j = 0; j < 2; ++j)
          acc[i][j] = wmma_bf16(af[i], bfr[j], acc[i][j]);
    }
  };

#if HAVE_ASYNC_LDS
  // ---- async double-buffer pipeline -------------------------------------
  auto stage = [&](int buf, int k0) {
#pragma unroll
    for (int i = 0; i < 2; ++i) {                  // A: 512 x 16B chunks
      int c = tid + i * 256, r = c >> 3, cc = (c & 7) * 8;
      cp_async_b128(&ldsA[buf][r * GEMM_BK + cc],
                    &A[(size_t)(m0 + r) * K + k0 + cc]);
    }
#pragma unroll
    for (int i = 0; i < 4; ++i) {                  // W: 1024 x 16B chunks
      int c = tid + i * 256, r = c >> 3, cc = (c & 7) * 8;
      cp_async_b128(&ldsB[buf][r * GEMM_BK + cc],
                    &W[(size_t)(n0 + r) * K + k0 + cc]);
    }
  };
  stage(0, 0);
  cp_async_wait0();
  __syncthreads();
  int cur = 0;
  for (int k0 = 0; k0 < K; k0 += GEMM_BK) {
    if (k0 + GEMM_BK < K) stage(cur ^ 1, k0 + GEMM_BK);
    compute(cur);
    cp_async_wait0();     // my async writes into the other buffer landed
    __syncthreads();      // everyone done reading cur; all writes visible
    cur ^= 1;
  }
#else
  // ---- register-prefetch double-buffer pipeline -------------------------
  v8bf ra[2], rb[4];
  auto loadRegs = [&](int k0) {
#pragma unroll
    for (int i = 0; i < 2; ++i) {
      int c = tid + i * 256, r = c >> 3, cc = (c & 7) * 8;
      ra[i] = *(const v8bf*)&A[(size_t)(m0 + r) * K + k0 + cc];
    }
#pragma unroll
    for (int i = 0; i < 4; ++i) {
      int c = tid + i * 256, r = c >> 3, cc = (c & 7) * 8;
      rb[i] = *(const v8bf*)&W[(size_t)(n0 + r) * K + k0 + cc];
    }
  };
  auto storeRegs = [&](int buf) {
#pragma unroll
    for (int i = 0; i < 2; ++i) {
      int c = tid + i * 256, r = c >> 3, cc = (c & 7) * 8;
      *(v8bf*)&ldsA[buf][r * GEMM_BK + cc] = ra[i];
    }
#pragma unroll
    for (int i = 0; i < 4; ++i) {
      int c = tid + i * 256, r = c >> 3, cc = (c & 7) * 8;
      *(v8bf*)&ldsB[buf][r * GEMM_BK + cc] = rb[i];
    }
  };
  loadRegs(0);
  storeRegs(0);
  __syncthreads();
  int cur = 0;
  for (int k0 = 0; k0 < K; k0 += GEMM_BK) {
    const bool nxt = (k0 + GEMM_BK) < K;
    if (nxt) loadRegs(k0 + GEMM_BK);   // global loads in flight...
    compute(cur);                      // ...hidden behind 8 WMMAs
    if (nxt) storeRegs(cur ^ 1);
    __syncthreads();
    cur ^= 1;
  }
#endif

  // Epilogue: C/D layout — VGPR r: M = r + 8*half, N = lane&15
  const int lane = tid & 31, half = lane >> 4, col = lane & 15;
#pragma unroll
  for (int i = 0; i < 2; ++i) {
#pragma unroll
    for (int j = 0; j < 2; ++j) {
#pragma unroll
      for (int r = 0; r < 8; ++r) {
        int m = m0 + wm * 32 + i * 16 + r + 8 * half;
        int n = n0 + wn * 32 + j * 16 + col;
        float v = acc[i][j][r];
        if (mode == 0) {
          if (bias) v += bias[n];
          outF[(size_t)m * N + n] = v;
        } else {
          int b = m / Sx, s = m % Sx;
          int h = n / DKx, dk = n % DKx;
          size_t idx = (mode == 1)
              ? (((size_t)(b * Hx + h) * Sx + s) * DKx + dk)
              : (((size_t)(b * Hx + h) * DKx + dk) * Sx + s);
          outB[idx] = (bf16)v;
        }
      }
    }
  }
}

// ---------------------------------------------------------------------------
// scores[b,h,q,k] = (Q·K^T)/8 + bias_table[q-k+MAXSEQ-1, h]  (raw, pre-softmax)
// Per (b,h): M=N=2048, K=64. Block tile 64x128, wave tile 32x32, direct global.
// ---------------------------------------------------------------------------
#define SC_BM 64
#define SC_BN 128

__global__ __launch_bounds__(256)
void attn_scores_kernel(const bf16* __restrict__ Q, const bf16* __restrict__ Kb,
                        const float* __restrict__ bias_table,
                        float* __restrict__ attn) {
  const int bh  = blockIdx.z;
  const int h   = bh % Hx;
  const int tid = threadIdx.x, wave = tid >> 5, wm = wave & 1, wn = wave >> 1;
  const int q0  = blockIdx.y * SC_BM;
  const int n0  = blockIdx.x * SC_BN;
  const bf16* Qh = Q  + (size_t)bh * Sx * DKx;
  const bf16* Kh = Kb + (size_t)bh * Sx * DKx;

  v8f acc[2][2] = {};
#pragma unroll
  for (int ks = 0; ks < DKx; ks += 32) {
    v16bf af[2], bfr[2];
#pragma unroll
    for (int i = 0; i < 2; ++i)
      af[i] = load_frag_A(Qh + (size_t)(q0 + wm * 32 + i * 16) * DKx + ks, DKx);
#pragma unroll
    for (int j = 0; j < 2; ++j)
      bfr[j] = load_frag_B_nk(Kh + (size_t)(n0 + wn * 32 + j * 16) * DKx + ks,
                              DKx);
#pragma unroll
    for (int i = 0; i < 2; ++i)
#pragma unroll
      for (int j = 0; j < 2; ++j)
        acc[i][j] = wmma_bf16(af[i], bfr[j], acc[i][j]);
  }

  const int lane = tid & 31, half = lane >> 4, col = lane & 15;
  const float scale = 0.125f;  // 1/sqrt(64)
#pragma unroll
  for (int i = 0; i < 2; ++i)
#pragma unroll
    for (int j = 0; j < 2; ++j)
#pragma unroll
      for (int r = 0; r < 8; ++r) {
        int q = q0 + wm * 32 + i * 16 + r + 8 * half;
        int k = n0 + wn * 32 + j * 16 + col;
        float v = acc[i][j][r] * scale
                + bias_table[(size_t)(q - k + MAXSEQ - 1) * Hx + h];
        attn[((size_t)bh * Sx + q) * Sx + k] = v;
      }
}

// ---------------------------------------------------------------------------
// In-place row softmax over attn rows of length S. One 256-thread block/row.
// ---------------------------------------------------------------------------
__global__ __launch_bounds__(256)
void softmax_rows_kernel(float* __restrict__ attn) {
  float* p = attn + (size_t)blockIdx.x * Sx;
  const int tid = threadIdx.x;
  __shared__ float red[8];

  float vals[8];
  float mx = -3.0e38f;
#pragma unroll
  for (int i = 0; i < 8; ++i) {
    vals[i] = p[tid + i * 256];
    mx = fmaxf(mx, vals[i]);
  }
#pragma unroll
  for (int off = 16; off; off >>= 1) mx = fmaxf(mx, __shfl_xor(mx, off, 32));
  if ((tid & 31) == 0) red[tid >> 5] = mx;
  __syncthreads();
  float m = red[0];
#pragma unroll
  for (int i = 1; i < 8; ++i) m = fmaxf(m, red[i]);
  __syncthreads();

  float sum = 0.0f;
#pragma unroll
  for (int i = 0; i < 8; ++i) {
    vals[i] = __expf(vals[i] - m);
    sum += vals[i];
  }
#pragma unroll
  for (int off = 16; off; off >>= 1) sum += __shfl_xor(sum, off, 32);
  if ((tid & 31) == 0) red[tid >> 5] = sum;
  __syncthreads();
  float s = 0.0f;
#pragma unroll
  for (int i = 0; i < 8; ++i) s += red[i];
  float inv = 1.0f / s;
#pragma unroll
  for (int i = 0; i < 8; ++i) p[tid + i * 256] = vals[i] * inv;
}

// ---------------------------------------------------------------------------
// context[b,s,h*DK+dk] = sum_k attn[b,h,s,k] * V[b,h,k,dk]
// Vt stored [B,H,DK,S] -> contiguous global B-frag loads.
// attn f32 tiles converted to bf16 while staging into double-buffered LDS;
// the next tile's global loads overlap the current tile's WMMAs.
// Block tile M=128 (8 waves x 16 rows), N=64 (4 n-tiles/wave), BK=32.
// ---------------------------------------------------------------------------
#define CT_BM 128

__global__ __launch_bounds__(256)
void attn_context_kernel(const float* __restrict__ attn,
                         const bf16* __restrict__ Vt,
                         bf16* __restrict__ ctx) {
  const int bh  = blockIdx.y;
  const int b   = bh / Hx, h = bh % Hx;
  const int m0  = blockIdx.x * CT_BM;
  const int tid = threadIdx.x, wave = tid >> 5;
  __shared__ alignas(16) bf16 ldsA[2][CT_BM * 32];   // 2 x 8 KB, [m][k]

  const float* Ah = attn + (size_t)bh * Sx * Sx;
  const bf16*  Vh = Vt   + (size_t)bh * DKx * Sx;

  v8f acc[4] = {};
  v4f rf[4];

  auto loadRegs = [&](int k0) {
#pragma unroll
    for (int i = 0; i < 4; ++i) {                  // 1024 float4 chunks
      int c = tid + i * 256, r = c >> 3, cc = (c & 7) * 4;
      rf[i] = *(const v4f*)&Ah[(size_t)(m0 + r) * Sx + k0 + cc];
    }
  };
  auto storeRegs = [&](int buf) {
#pragma unroll
    for (int i = 0; i < 4; ++i) {
      int c = tid + i * 256, r = c >> 3, cc = (c & 7) * 4;
      v4bf hv;
      hv[0] = (bf16)rf[i][0]; hv[1] = (bf16)rf[i][1];
      hv[2] = (bf16)rf[i][2]; hv[3] = (bf16)rf[i][3];
      *(v4bf*)&ldsA[buf][r * 32 + cc] = hv;        // single 8B LDS store
    }
  };

  loadRegs(0);
  storeRegs(0);
  __syncthreads();
  int cur = 0;
  for (int k0 = 0; k0 < Sx; k0 += 32) {
    const bool nxt = (k0 + 32) < Sx;
    if (nxt) loadRegs(k0 + 32);

    v16bf afrag = load_frag_A(&ldsA[cur][wave * 16 * 32], 32);
#pragma unroll
    for (int j = 0; j < 4; ++j) {
      v16bf bfrag = load_frag_B_nk(Vh + (size_t)(j * 16) * Sx + k0, Sx);
      acc[j] = wmma_bf16(afrag, bfrag, acc[j]);
    }

    if (nxt) storeRegs(cur ^ 1);
    __syncthreads();
    cur ^= 1;
  }

  const int lane = tid & 31, half = lane >> 4, col = lane & 15;
#pragma unroll
  for (int j = 0; j < 4; ++j)
#pragma unroll
    for (int r = 0; r < 8; ++r) {
      int s = m0 + wave * 16 + r + 8 * half;
      int n = h * DKx + j * 16 + col;
      ctx[(size_t)(b * Sx + s) * Dx + n] = (bf16)acc[j][r];
    }
}

// ---------------------------------------------------------------------------
// Host-side launch
// ---------------------------------------------------------------------------
extern "C" void kernel_launch(void* const* d_in, const int* in_sizes, int n_in,
                              void* d_out, int out_size, void* d_ws, size_t ws_size,
                              hipStream_t stream) {
  const float* q_f32   = (const float*)d_in[0];
  const float* k_f32   = (const float*)d_in[1];
  const float* v_f32   = (const float*)d_in[2];
  const float* Wq_f32  = (const float*)d_in[3];
  const float* Wk_f32  = (const float*)d_in[4];
  const float* Wv_f32  = (const float*)d_in[5];
  const float* Wo_f32  = (const float*)d_in[6];
  const float* bo      = (const float*)d_in[7];
  const float* biasTab = (const float*)d_in[8];

  float* out  = (float*)d_out;                       // [B,S,D]
  float* attn = out + (size_t)Bx * Sx * Dx;          // [B,H,S,S]

  // Workspace layout (all segments are multiples of 256 B); total 64 MB.
  const size_t nBSD = (size_t)Bx * Sx * Dx;          // 4,194,304
  const size_t nDD  = (size_t)Dx * Dx;               // 1,048,576
  char* ws = (char*)d_ws;
  size_t off = 0;
  bf16* Xq  = (bf16*)(ws + off); off += nBSD * 2;
  bf16* Xk  = (bf16*)(ws + off); off += nBSD * 2;
  bf16* Xv  = (bf16*)(ws + off); off += nBSD * 2;
  bf16* Wqb = (bf16*)(ws + off); off += nDD * 2;
  bf16* Wkb = (bf16*)(ws + off); off += nDD * 2;
  bf16* Wvb = (bf16*)(ws + off); off += nDD * 2;
  bf16* Wob = (bf16*)(ws + off); off += nDD * 2;
  bf16* Qb  = (bf16*)(ws + off); off += nBSD * 2;    // [B,H,S,DK]
  bf16* Kb  = (bf16*)(ws + off); off += nBSD * 2;    // [B,H,S,DK]
  bf16* Vtb = (bf16*)(ws + off); off += nBSD * 2;    // [B,H,DK,S]
  bf16* Ctx = (bf16*)(ws + off); off += nBSD * 2;    // [B*S, D]

  // 1) f32 -> bf16 conversions
  f32_to_bf16_kernel<<<2048, 256, 0, stream>>>(q_f32,  Xq,  nBSD);
  f32_to_bf16_kernel<<<2048, 256, 0, stream>>>(k_f32,  Xk,  nBSD);
  f32_to_bf16_kernel<<<2048, 256, 0, stream>>>(v_f32,  Xv,  nBSD);
  f32_to_bf16_kernel<<<1024, 256, 0, stream>>>(Wq_f32, Wqb, nDD);
  f32_to_bf16_kernel<<<1024, 256, 0, stream>>>(Wk_f32, Wkb, nDD);
  f32_to_bf16_kernel<<<1024, 256, 0, stream>>>(Wv_f32, Wvb, nDD);
  f32_to_bf16_kernel<<<1024, 256, 0, stream>>>(Wo_f32, Wob, nDD);

  // 2) QKV projections: [4096,1024] x [1024,1024]^T
  dim3 ggemm(Dx / GEMM_BN, (Bx * Sx) / GEMM_BM);     // (8, 64)
  gemm_bf16_nt<<<ggemm, 256, 0, stream>>>(Xq, Wqb, nullptr, nullptr, Qb,
                                          Bx * Sx, Dx, Dx, /*mode=*/1);
  gemm_bf16_nt<<<ggemm, 256, 0, stream>>>(Xk, Wkb, nullptr, nullptr, Kb,
                                          Bx * Sx, Dx, Dx, /*mode=*/1);
  gemm_bf16_nt<<<ggemm, 256, 0, stream>>>(Xv, Wvb, nullptr, nullptr, Vtb,
                                          Bx * Sx, Dx, Dx, /*mode=*/2);

  // 3) raw scores + bias -> attn region of d_out
  dim3 gsc(Sx / SC_BN, Sx / SC_BM, Bx * Hx);         // (16, 32, 32)
  attn_scores_kernel<<<gsc, 256, 0, stream>>>(Qb, Kb, biasTab, attn);

  // 4) softmax in place over rows
  softmax_rows_kernel<<<dim3(Bx * Hx * Sx), 256, 0, stream>>>(attn);

  // 5) context = attn @ V
  dim3 gct(Sx / CT_BM, Bx * Hx);                     // (16, 32)
  attn_context_kernel<<<gct, 256, 0, stream>>>(attn, Vtb, Ctx);

  // 6) output projection + bias, f32 out
  gemm_bf16_nt<<<ggemm, 256, 0, stream>>>(Ctx, Wob, bo, out, nullptr,
                                          Bx * Sx, Dx, Dx, /*mode=*/0);
}
